// encoder_decoder_model_82248623719117
// MI455X (gfx1250) — matrix operand order repaired and checked by reading, not compiled, not run on hardware
//
#include <hip/hip_runtime.h>

typedef unsigned short u16;
typedef __attribute__((ext_vector_type(16))) __bf16 bf16x16;
typedef __attribute__((ext_vector_type(8)))  float  f32x8;
typedef __attribute__((ext_vector_type(4)))  unsigned int u32x4;

// ---------------- model dims ----------------
constexpr int kB = 32;        // batch
constexpr int kT = 64;        // encoder steps
constexpr int kJ = 64;        // decoder steps
constexpr int kE = 512;       // embed dim
constexpr int kH = 1024;      // hidden dim
constexpr int kV = 32000;     // vocab
constexpr int k4H = 4 * kH;   // gate width

// ---------------- helpers ----------------
__device__ __forceinline__ u16 f2bf(float f) {
  unsigned int x = __float_as_uint(f);
  unsigned int r = x + 0x7fffu + ((x >> 16) & 1u);   // round-to-nearest-even
  return (u16)(r >> 16);
}
__device__ __forceinline__ float sigmoidf_(float x) {
  return 1.0f / (1.0f + expf(-x));
}

struct Frag32B { u32x4 lo, hi; };

// A-matrix 16x32 bf16 fragment (row = per-lane M row pointer, K-contiguous memory)
// lanes 0-15 : K 0-7 (VGPR0-3) and K 16-23 (VGPR4-7)
// lanes 16-31: K 8-15            and K 24-31
__device__ __forceinline__ bf16x16 load_a_frag(const u16* __restrict__ row, int kk, int half) {
  Frag32B f;
  f.lo = *(const u32x4*)(row + kk + 8 * half);
  f.hi = *(const u32x4*)(row + kk + 16 + 8 * half);
  return __builtin_bit_cast(bf16x16, f);
}
// B-matrix 32x16 bf16 fragment (row = W[col n] row pointer, K-contiguous)
// lanes 0-15 hold K 0-7 and K 16-23, lanes 16-31 hold K 8-15 and K 24-31
__device__ __forceinline__ bf16x16 load_b_frag(const u16* __restrict__ row, int kk, int half) {
  Frag32B f;
  f.lo = *(const u32x4*)(row + kk + 8 * half);
  f.hi = *(const u32x4*)(row + kk + 16 + 8 * half);
  return __builtin_bit_cast(bf16x16, f);
}
__device__ __forceinline__ f32x8 wmma_bf16(bf16x16 a, bf16x16 b, f32x8 c) {
  return __builtin_amdgcn_wmma_f32_16x16x32_bf16(false, a, false, b, (short)0, c, false, false);
}

// ---------------- small utility kernels ----------------
__global__ void f32_to_bf16_kernel(const float* __restrict__ src, u16* __restrict__ dst, long n) {
  long i = (long)blockIdx.x * blockDim.x + threadIdx.x;
  long stride = (long)gridDim.x * blockDim.x;
  for (; i < n; i += stride) dst[i] = f2bf(src[i]);
}
__global__ void f32_copy_kernel(const float* __restrict__ src, float* __restrict__ dst, long n) {
  long i = (long)blockIdx.x * blockDim.x + threadIdx.x;
  long stride = (long)gridDim.x * blockDim.x;
  for (; i < n; i += stride) dst[i] = src[i];
}
// dst[t][b][e] = bf16(emb[tok[b][t]][e]);  tok layout [B, steps]
__global__ void embed_gather_kernel(const int* __restrict__ tok, const float* __restrict__ emb,
                                    u16* __restrict__ dst, int steps, long total) {
  long i = (long)blockIdx.x * blockDim.x + threadIdx.x;
  long stride = (long)gridDim.x * blockDim.x;
  for (; i < total; i += stride) {
    int t = (int)(i / (kB * kE));
    long rem = i % (kB * kE);
    int b = (int)(rem / kE);
    int e = (int)(rem % kE);
    int tk = tok[b * steps + t];
    dst[i] = f2bf(emb[(size_t)tk * kE + e]);
  }
}
__global__ void len_kernel(const int* __restrict__ mask, int* __restrict__ len) {
  int b = threadIdx.x;
  if (b < kB) {
    int s = 0;
    for (int t = 0; t < kT; ++t) s += mask[b * kT + t];
    len[b] = s - 1;
  }
}

// ---------------- fused LSTM step ----------------
// grid.x = H/16 column tiles; block = 128 (4 waves, one per gate i/f/g/o).
// Each wave: 2 M-tiles (batch rows 0-15, 16-31) x 1 N-tile (16 cols) over K = E + H.
__global__ __launch_bounds__(128)
void lstm_step_kernel(const u16* __restrict__ x_t,    // [B][E] bf16
                      const u16* __restrict__ h_in,   // [B][H] bf16
                      const u16* __restrict__ Wih,    // [4H][E] bf16
                      const u16* __restrict__ Whh,    // [4H][H] bf16
                      const float* __restrict__ bih,  // [4H]
                      const float* __restrict__ bhh,  // [4H]
                      float* __restrict__ cbuf,       // [B][H] f32, updated in place
                      u16* __restrict__ h_out,        // [B][H] bf16 (next-step input)
                      float* __restrict__ hid_f32,    // nullable: d_out hiddens + j*H
                      u16* __restrict__ hid_bf,       // nullable: hiddens bf16 + j*H
                      const int* __restrict__ len,    // nullable: encoder final-state select
                      int t,
                      float* __restrict__ h_sel,      // [B][H] f32 (encoder gather)
                      float* __restrict__ c_sel) {    // [B][H] f32
  __shared__ float lds[4][32][16];

  const int lane = threadIdx.x & 31;
  const int gate = threadIdx.x >> 5;     // 0=i 1=f 2=g 3=o
  const int n    = lane & 15;
  const int half = lane >> 4;
  const int colbase = blockIdx.x * 16;
  const int mrow = lane & 15;

  const u16* arow0_x = x_t  + (size_t)(mrow)      * kE;
  const u16* arow1_x = x_t  + (size_t)(16 + mrow) * kE;
  const u16* arow0_h = h_in + (size_t)(mrow)      * kH;
  const u16* arow1_h = h_in + (size_t)(16 + mrow) * kH;
  const u16* brow_x  = Wih  + (size_t)(gate * kH + colbase + n) * kE;
  const u16* brow_h  = Whh  + (size_t)(gate * kH + colbase + n) * kH;

  f32x8 acc0 = {};
  f32x8 acc1 = {};

  // K over E (input contribution)
#pragma unroll 4
  for (int kk = 0; kk < kE; kk += 32) {
    bf16x16 a0 = load_a_frag(arow0_x, kk, half);
    bf16x16 a1 = load_a_frag(arow1_x, kk, half);
    bf16x16 bb = load_b_frag(brow_x, kk, half);
    acc0 = wmma_bf16(a0, bb, acc0);
    acc1 = wmma_bf16(a1, bb, acc1);
  }
  // K over H (recurrent contribution)
#pragma unroll 4
  for (int kk = 0; kk < kH; kk += 32) {
    bf16x16 a0 = load_a_frag(arow0_h, kk, half);
    bf16x16 a1 = load_a_frag(arow1_h, kk, half);
    bf16x16 bb = load_b_frag(brow_h, kk, half);
    acc0 = wmma_bf16(a0, bb, acc0);
    acc1 = wmma_bf16(a1, bb, acc1);
  }

  const float bias = bih[gate * kH + colbase + n] + bhh[gate * kH + colbase + n];
#pragma unroll
  for (int v = 0; v < 8; ++v) {
    lds[gate][v + 8 * half][n]      = acc0[v] + bias;   // M = v + 8*half
    lds[gate][16 + v + 8 * half][n] = acc1[v] + bias;   // M = 16 + v + 8*half
  }
  __syncthreads();

  // 512 gate elements for this column tile; 4 per thread
  for (int e = threadIdx.x; e < 512; e += 128) {
    int m  = e >> 4;
    int nn = e & 15;
    float ig = sigmoidf_(lds[0][m][nn]);
    float fg = sigmoidf_(lds[1][m][nn]);
    float gg = tanhf(lds[2][m][nn]);
    float og = sigmoidf_(lds[3][m][nn]);
    size_t cidx = (size_t)m * kH + colbase + nn;
    float cn = fg * cbuf[cidx] + ig * gg;
    float hn = og * tanhf(cn);
    cbuf[cidx]  = cn;
    h_out[cidx] = f2bf(hn);
    if (hid_f32) {
      size_t hidx = (size_t)m * (kJ * kH) + colbase + nn;   // row b*J + j
      hid_f32[hidx] = hn;
      hid_bf[hidx]  = f2bf(hn);
    }
    if (len && len[m] == t) {
      h_sel[cidx] = hn;
      c_sel[cidx] = cn;
    }
  }
}

// ---------------- projection GEMM: [2048,1024]bf16 x [32000,1024]^T bf16 -> f32 logits ----------------
// grid = (V/256, M/128); block = 256 (8 waves).
// Wave w: m-group = w&1 (4 m-tiles = 64 rows), n-group = w>>1 (4 n-tiles = 64 cols).
// 4x4 register blocking: 16 WMMAs per 8 fragment loads per K-step (1 WMMA per b128 load).
__global__ __launch_bounds__(256)
void proj_kernel(const u16* __restrict__ X,     // [2048][1024] bf16
                 const u16* __restrict__ W,     // [32000][1024] bf16
                 const float* __restrict__ bias,// [32000]
                 float* __restrict__ out) {     // [2048][32000] f32
  const int lane = threadIdx.x & 31;
  const int wave = threadIdx.x >> 5;            // 0..7
  const int n    = lane & 15;
  const int half = lane >> 4;
  const int m0 = blockIdx.y * 128 + (wave & 1) * 64;
  const int n0 = blockIdx.x * 256 + (wave >> 1) * 64;

  const u16* arow[4];
  const u16* brow[4];
#pragma unroll
  for (int i = 0; i < 4; ++i) arow[i] = X + (size_t)(m0 + i * 16 + (lane & 15)) * kH;
#pragma unroll
  for (int j = 0; j < 4; ++j) brow[j] = W + (size_t)(n0 + j * 16 + n) * kH;

  f32x8 acc[4][4] = {};

#pragma unroll 2
  for (int kk = 0; kk < kH; kk += 32) {
    bf16x16 a[4];
#pragma unroll
    for (int i = 0; i < 4; ++i) a[i] = load_a_frag(arow[i], kk, half);
#pragma unroll
    for (int j = 0; j < 4; ++j) {
      bf16x16 bb = load_b_frag(brow[j], kk, half);
#pragma unroll
      for (int i = 0; i < 4; ++i) acc[i][j] = wmma_bf16(a[i], bb, acc[i][j]);
    }
  }

#pragma unroll
  for (int j = 0; j < 4; ++j) {
    const int col = n0 + j * 16 + n;
    const float bc = bias[col];
#pragma unroll
    for (int i = 0; i < 4; ++i) {
#pragma unroll
      for (int v = 0; v < 8; ++v) {
        size_t mrow = (size_t)(m0 + i * 16 + v + 8 * half);
        out[mrow * kV + col] = acc[i][j][v] + bc;
      }
    }
  }
}

// ---------------- in-place log_softmax over rows of [2048][32000] ----------------
__global__ __launch_bounds__(256)
void logsoftmax_kernel(float* __restrict__ out) {
  float* row = out + (size_t)blockIdx.x * kV;
  __shared__ float red[8];
  __shared__ float bc;
  const int tid = threadIdx.x, lane = tid & 31, wv = tid >> 5;

  float mx = -3.4e38f;
  for (int i = tid; i < kV; i += 256) mx = fmaxf(mx, row[i]);
  for (int m = 16; m; m >>= 1) mx = fmaxf(mx, __shfl_xor(mx, m, 32));
  if (lane == 0) red[wv] = mx;
  __syncthreads();
  if (tid == 0) {
    float r = red[0];
    for (int i = 1; i < 8; ++i) r = fmaxf(r, red[i]);
    bc = r;
  }
  __syncthreads();
  mx = bc;

  float s = 0.f;
  for (int i = tid; i < kV; i += 256) s += expf(row[i] - mx);
  for (int m = 16; m; m >>= 1) s += __shfl_xor(s, m, 32);
  if (lane == 0) red[wv] = s;
  __syncthreads();
  if (tid == 0) {
    float r = 0.f;
    for (int i = 0; i < 8; ++i) r += red[i];
    bc = logf(r);
  }
  __syncthreads();
  const float lse = bc;
  for (int i = tid; i < kV; i += 256) row[i] = row[i] - mx - lse;
}

// ---------------- host orchestration ----------------
static inline int cvt_blocks(long n) {
  long b = (n + 255) / 256;
  return (int)(b > 262144 ? 262144 : b);
}

extern "C" void kernel_launch(void* const* d_in, const int* in_sizes, int n_in,
                              void* d_out, int out_size, void* d_ws, size_t ws_size,
                              hipStream_t stream) {
  const int*   eng  = (const int*)d_in[0];
  const int*   mask = (const int*)d_in[1];
  const int*   zh   = (const int*)d_in[2];
  const float* h0   = (const float*)d_in[3];
  const float* c0   = (const float*)d_in[4];
  const float* embE = (const float*)d_in[5];
  const float* embC = (const float*)d_in[6];
  const float* eWih = (const float*)d_in[7];
  const float* eWhh = (const float*)d_in[8];
  const float* ebih = (const float*)d_in[9];
  const float* ebhh = (const float*)d_in[10];
  const float* dWih = (const float*)d_in[11];
  const float* dWhh = (const float*)d_in[12];
  const float* dbih = (const float*)d_in[13];
  const float* dbhh = (const float*)d_in[14];
  const float* linW = (const float*)d_in[15];
  const float* linb = (const float*)d_in[16];
  float* out = (float*)d_out;

  // workspace carve-up (256B aligned)
  char* ws = (char*)d_ws;
  size_t o = 0;
  auto carve = [&](size_t bytes) { void* p = ws + o; o += (bytes + 255) & ~(size_t)255; return p; };
  u16*   x_enc = (u16*)  carve((size_t)kT * kB * kE * 2);
  u16*   x_dec = (u16*)  carve((size_t)kJ * kB * kE * 2);
  u16*   WihE  = (u16*)  carve((size_t)k4H * kE * 2);
  u16*   WhhE  = (u16*)  carve((size_t)k4H * kH * 2);
  u16*   WihD  = (u16*)  carve((size_t)k4H * kE * 2);
  u16*   WhhD  = (u16*)  carve((size_t)k4H * kH * 2);
  u16*   Wlin  = (u16*)  carve((size_t)kV * kH * 2);
  u16*   hbf0  = (u16*)  carve((size_t)kB * kH * 2);
  u16*   hbf1  = (u16*)  carve((size_t)kB * kH * 2);
  float* cbuf  = (float*)carve((size_t)kB * kH * 4);
  float* hsel  = (float*)carve((size_t)kB * kH * 4);
  float* csel  = (float*)carve((size_t)kB * kH * 4);
  u16*   hidbf = (u16*)  carve((size_t)kB * kJ * kH * 2);
  int*   len   = (int*)  carve(kB * 4);
  u16*   hbf[2] = { hbf0, hbf1 };

  // weight / input conversion to bf16
  f32_to_bf16_kernel<<<cvt_blocks((long)k4H * kE), 256, 0, stream>>>(eWih, WihE, (long)k4H * kE);
  f32_to_bf16_kernel<<<cvt_blocks((long)k4H * kH), 256, 0, stream>>>(eWhh, WhhE, (long)k4H * kH);
  f32_to_bf16_kernel<<<cvt_blocks((long)k4H * kE), 256, 0, stream>>>(dWih, WihD, (long)k4H * kE);
  f32_to_bf16_kernel<<<cvt_blocks((long)k4H * kH), 256, 0, stream>>>(dWhh, WhhD, (long)k4H * kH);
  f32_to_bf16_kernel<<<cvt_blocks((long)kV * kH), 256, 0, stream>>>(linW, Wlin, (long)kV * kH);
  embed_gather_kernel<<<cvt_blocks((long)kT * kB * kE), 256, 0, stream>>>(eng, embE, x_enc, kT, (long)kT * kB * kE);
  embed_gather_kernel<<<cvt_blocks((long)kJ * kB * kE), 256, 0, stream>>>(zh, embC, x_dec, kJ, (long)kJ * kB * kE);
  len_kernel<<<1, 32, 0, stream>>>(mask, len);
  f32_to_bf16_kernel<<<cvt_blocks((long)kB * kH), 256, 0, stream>>>(h0, hbf0, (long)kB * kH);
  f32_copy_kernel<<<cvt_blocks((long)kB * kH), 256, 0, stream>>>(c0, cbuf, (long)kB * kH);

  // encoder scan (h double-buffered in bf16; c in place; final state gathered by mask length)
  for (int t = 0; t < kT; ++t) {
    lstm_step_kernel<<<kH / 16, 128, 0, stream>>>(
        x_enc + (size_t)t * kB * kE, hbf[t & 1], WihE, WhhE, ebih, ebhh,
        cbuf, hbf[(t + 1) & 1],
        nullptr, nullptr, len, t, hsel, csel);
  }

  // decoder init from gathered encoder state
  f32_to_bf16_kernel<<<cvt_blocks((long)kB * kH), 256, 0, stream>>>(hsel, hbf0, (long)kB * kH);
  f32_copy_kernel<<<cvt_blocks((long)kB * kH), 256, 0, stream>>>(csel, cbuf, (long)kB * kH);

  // decoder scan; writes hiddens (f32 -> d_out tail region, bf16 -> projection input)
  float* hid_out = out + (size_t)kB * kJ * kV;
  for (int j = 0; j < kJ; ++j) {
    lstm_step_kernel<<<kH / 16, 128, 0, stream>>>(
        x_dec + (size_t)j * kB * kE, hbf[j & 1], WihD, WhhD, dbih, dbhh,
        cbuf, hbf[(j + 1) & 1],
        hid_out + (size_t)j * kH, hidbf + (size_t)j * kH,
        nullptr, 0, nullptr, nullptr);
  }

  // projection + log_softmax (logits written straight into d_out, normalized in place)
  proj_kernel<<<dim3(kV / 256, (kB * kJ) / 128), 256, 0, stream>>>(hidbf, Wlin, linb, out);
  logsoftmax_kernel<<<kB * kJ, 256, 0, stream>>>(out);
}
